// LanguageModel_61942018343264
// MI455X (gfx1250) — compile-verified, matrix operand order checked by reading, not run on hardware
//
#include <hip/hip_runtime.h>
#include <hip/hip_bf16.h>
#include <stdint.h>

// GPT-style forward for MI455X (gfx1250): bf16 WMMA GEMMs (async-LDS double-buffered)
// + fused flash attention.  V=32000 E=1024 T=2048 L=4 H=16 HD=64 B=2

#define V_SZ 32000
#define E_SZ 1024
#define TCTX 2048
#define L_N  4
#define H_N  16
#define HD_SZ 64
#define BATCH 2
#define BT (BATCH * TCTX)
#define EPSV 1e-5f

typedef __attribute__((ext_vector_type(16))) __bf16 v16bf;
typedef __attribute__((ext_vector_type(8)))  float  v8f;
typedef int v4i __attribute__((ext_vector_type(4)));
typedef unsigned short bfu;

union FragA { v16bf v; uint4 q[2]; bfu u[16]; };
union FragC { v8f   v; float f[8]; };

__device__ __forceinline__ bfu f2bfu(float f) {
  union { float f; uint32_t u; } p; p.f = f;
  uint32_t r = p.u + 0x7FFFu + ((p.u >> 16) & 1u);
  return (bfu)(r >> 16);
}

// ---------------- async global->LDS staging (CDNA5 ASYNCcnt path), with fallback
#if __has_builtin(__builtin_amdgcn_global_load_async_to_lds_b128)
#define USE_ASYNC_LDS 1
typedef __attribute__((address_space(1))) v4i* gv4i_p;
typedef __attribute__((address_space(3))) v4i* lv4i_p;
#endif

__device__ __forceinline__ void stage16(uint4* dst, const bfu* src) {
#ifdef USE_ASYNC_LDS
  __builtin_amdgcn_global_load_async_to_lds_b128(
      (gv4i_p)(v4i*)src, (lv4i_p)(v4i*)dst, 0, 0);
#else
  *dst = *(const uint4*)src;
#endif
}

__device__ __forceinline__ void wait_async() {
#ifdef USE_ASYNC_LDS
#if __has_builtin(__builtin_amdgcn_s_wait_asynccnt)
  __builtin_amdgcn_s_wait_asynccnt(0);
#else
  asm volatile("s_wait_asynccnt 0" ::: "memory");
#endif
#endif
}

// ---------------------------------------------------------------- embedding
__global__ void embed_kernel(const int* __restrict__ idx, const float* __restrict__ tok,
                             const float* __restrict__ pos, float* __restrict__ x) {
  int r = blockIdx.x;                 // [0, BT)
  int t = r % TCTX;
  int tk = idx[r];
  const float* te = tok + (size_t)tk * E_SZ;
  const float* pe = pos + (size_t)t  * E_SZ;
  float* xr = x + (size_t)r * E_SZ;
  for (int i = threadIdx.x; i < E_SZ; i += blockDim.x) xr[i] = te[i] + pe[i];
}

// ---------------------------------------------------------------- layernorm (f32 -> bf16)
__global__ void ln_kernel(const float* __restrict__ x, const float* __restrict__ g,
                          const float* __restrict__ b, bfu* __restrict__ h) {
  __shared__ float red[256];
  int r = blockIdx.x, tid = threadIdx.x;
  const float* xr = x + (size_t)r * E_SZ;
  float v0[4], s = 0.f;
  for (int i = 0; i < 4; i++) { v0[i] = xr[tid + i * 256]; s += v0[i]; }
  red[tid] = s; __syncthreads();
  for (int o = 128; o > 0; o >>= 1) { if (tid < o) red[tid] += red[tid + o]; __syncthreads(); }
  float mean = red[0] * (1.0f / E_SZ); __syncthreads();
  float s2 = 0.f;
  for (int i = 0; i < 4; i++) { float d = v0[i] - mean; s2 += d * d; }
  red[tid] = s2; __syncthreads();
  for (int o = 128; o > 0; o >>= 1) { if (tid < o) red[tid] += red[tid + o]; __syncthreads(); }
  float rstd = rsqrtf(red[0] * (1.0f / E_SZ) + EPSV);
  bfu* hr = h + (size_t)r * E_SZ;
  for (int i = 0; i < 4; i++) {
    int c = tid + i * 256;
    hr[c] = f2bfu((v0[i] - mean) * rstd * g[c] + b[c]);
  }
}

// ------------------------------------------- weight packing f32 -> bf16, TRANSPOSED [N,K]
// in: [K,N] row-major f32 ; out: [N,K] row-major bf16
__global__ void pack_wT_kernel(const float* __restrict__ w, bfu* __restrict__ out,
                               int K, int N) {
  size_t total = (size_t)K * N;
  size_t i = (size_t)blockIdx.x * blockDim.x + threadIdx.x;
  size_t stride = (size_t)gridDim.x * blockDim.x;
  for (; i < total; i += stride) {
    size_t n = i / (size_t)K;
    size_t k = i - n * (size_t)K;
    out[i] = f2bfu(w[k * (size_t)N + n]);
  }
}

// [H,E,HD] -> transposed [H*HD, E] (rows are head-concat output cols), f32 -> bf16
__global__ void pack_qkvT_kernel(const float* __restrict__ w, bfu* __restrict__ out) {
  int i = blockIdx.x * blockDim.x + threadIdx.x;  // grid covers E*E exactly
  int col = i >> 10;         // output row  = h*HD + d
  int e   = i & 1023;        // output col  = e (K index)
  int hh  = col >> 6;
  int d   = col & 63;
  out[i] = f2bfu(w[((size_t)hh * E_SZ + e) * HD_SZ + d]);
}

// ---------------------------------------------------------------- bf16 WMMA GEMM
// C[M,N] = A[M,K] (bf16 rm) x Bt[N,K] (bf16, pre-transposed), fp32 accum.
// Block tile 128x128x32, double-buffered async LDS staging; 8 waves x (64Mx32N).
template <bool BIAS, bool RELU, bool RESID, bool OUTBF16>
__global__ void __launch_bounds__(256)
gemm_bf16_kernel(const bfu* __restrict__ A, const bfu* __restrict__ Bt,
                 const float* __restrict__ bias, const float* __restrict__ resid,
                 void* __restrict__ outp, int M, int N, int K) {
  __shared__ uint4 As4[2][128 * 4];   // A tile 128x32 bf16
  __shared__ uint4 Bs4[2][128 * 4];   // B tile [n][k] 128x32 bf16

  const int tid  = threadIdx.x;
  const int lane = tid & 31;
  const int wave = tid >> 5;
  const int m0 = blockIdx.y * 128;
  const int n0 = blockIdx.x * 128;
  const int wm = wave >> 2;          // 0..1 : 64 rows
  const int wn = wave & 3;           // 0..3 : 32 cols
  const int c16 = lane & 15;
  const int hf  = lane >> 4;         // half-wave select
  const int kq  = hf;                // uint4 chunk base per ISA 16-bit layout (k 0/8)

  // this thread's 2 staging chunks: chunk ch -> (row, kc)
  const int ch0_row = (tid * 2) >> 2, ch0_kc = (tid * 2) & 3;
  const int ch1_row = (tid * 2 + 1) >> 2, ch1_kc = (tid * 2 + 1) & 3;

  FragC acc[4][2];
  for (int i = 0; i < 4; i++)
    for (int j = 0; j < 2; j++)
      for (int e = 0; e < 8; e++) acc[i][j].f[e] = 0.0f;

#define STAGE(buf, k0)                                                              \
  do {                                                                              \
    stage16(&As4[buf][ch0_row * 4 + ch0_kc],                                        \
            A + (size_t)(m0 + ch0_row) * K + (k0) + ch0_kc * 8);                    \
    stage16(&As4[buf][ch1_row * 4 + ch1_kc],                                        \
            A + (size_t)(m0 + ch1_row) * K + (k0) + ch1_kc * 8);                    \
    stage16(&Bs4[buf][ch0_row * 4 + ch0_kc],                                        \
            Bt + (size_t)(n0 + ch0_row) * K + (k0) + ch0_kc * 8);                   \
    stage16(&Bs4[buf][ch1_row * 4 + ch1_kc],                                        \
            Bt + (size_t)(n0 + ch1_row) * K + (k0) + ch1_kc * 8);                   \
  } while (0)

  STAGE(0, 0);
  for (int k0 = 0; k0 < K; k0 += 32) {
    const int cur = (k0 >> 5) & 1;
    wait_async();
    __syncthreads();                 // tile `cur` visible to all waves
    if (k0 + 32 < K) STAGE(cur ^ 1, k0 + 32);

    FragA aF[4], bF[2];
    for (int mi = 0; mi < 4; mi++) {
      int row = wm * 64 + mi * 16 + c16;
      aF[mi].q[0] = As4[cur][row * 4 + kq];
      aF[mi].q[1] = As4[cur][row * 4 + kq + 2];
    }
    for (int ni = 0; ni < 2; ni++) {
      int col = wn * 32 + ni * 16 + c16;
      bF[ni].q[0] = Bs4[cur][col * 4 + kq];
      bF[ni].q[1] = Bs4[cur][col * 4 + kq + 2];
    }
    for (int mi = 0; mi < 4; mi++)
      for (int ni = 0; ni < 2; ni++)
        acc[mi][ni].v = __builtin_amdgcn_wmma_f32_16x16x32_bf16(
            false, aF[mi].v, false, bF[ni].v, (short)0, acc[mi][ni].v, false, false);
  }
#undef STAGE

  // epilogue: C element (j + 8*hf, c16) per VGPR j
  for (int mi = 0; mi < 4; mi++)
    for (int ni = 0; ni < 2; ni++) {
      int colg = n0 + wn * 32 + ni * 16 + c16;
      float bi = 0.0f;
      if (BIAS) bi = bias[colg];
      for (int j = 0; j < 8; j++) {
        int rowg = m0 + wm * 64 + mi * 16 + j + 8 * hf;
        float v = acc[mi][ni].f[j] + bi;
        if (RESID) v += resid[(size_t)rowg * N + colg];
        if (RELU)  v = fmaxf(v, 0.0f);
        if (OUTBF16) ((bfu*)outp)[(size_t)rowg * N + colg] = f2bfu(v);
        else         ((float*)outp)[(size_t)rowg * N + colg] = v;
      }
    }
}

// ---------------------------------------------------------------- fused causal flash attention
// q/k/v/out: [B*T, E] bf16, columns grouped by head (h*64+d). One wave = 16 query rows.
__global__ void __launch_bounds__(256)
attention_kernel(const bfu* __restrict__ qb, const bfu* __restrict__ kb,
                 const bfu* __restrict__ vb, bfu* __restrict__ ab) {
  __shared__ uint4 Pb4[8][16 * 4];   // per-wave P tile 16x32 bf16
  const int blk    = blockIdx.x;     // B * H * (T/128)
  const int qchunk = blk % (TCTX / 128);
  const int h      = (blk / (TCTX / 128)) % H_N;
  const int b      = blk / ((TCTX / 128) * H_N);
  const int wave = threadIdx.x >> 5;
  const int lane = threadIdx.x & 31;
  const int r16  = lane & 15;
  const int hf   = lane >> 4;
  const int ka   = hf * 8;
  const int q0   = qchunk * 128 + wave * 16;
  bfu* Pbe = (bfu*)Pb4[wave];

  // Q fragments (row = q0+r16, d split in two K=32 steps)
  FragA aq[2];
  {
    const bfu* qrow = qb + ((size_t)(b * TCTX + q0 + r16)) * E_SZ + h * HD_SZ;
    for (int f = 0; f < 2; f++) {
      aq[f].q[0] = *(const uint4*)(qrow + f * 32 + ka);
      aq[f].q[1] = *(const uint4*)(qrow + f * 32 + ka + 16);
    }
  }

  float mrow[8], lrow[8];
  FragC O[4];
  for (int j = 0; j < 8; j++) { mrow[j] = -1e30f; lrow[j] = 0.0f; }
  for (int n = 0; n < 4; n++) for (int j = 0; j < 8; j++) O[n].f[j] = 0.0f;
  const float scale = 0.125f;  // HD^-0.5

  for (int kt = 0; kt <= q0 + 15; kt += 32) {
    FragC S0, S1;
    for (int j = 0; j < 8; j++) { S0.f[j] = 0.f; S1.f[j] = 0.f; }
    for (int f = 0; f < 2; f++) {
      FragA bk;
      const bfu* krow = kb + ((size_t)(b * TCTX + kt + r16)) * E_SZ + h * HD_SZ;
      bk.q[0] = *(const uint4*)(krow + f * 32 + ka);
      bk.q[1] = *(const uint4*)(krow + f * 32 + ka + 16);
      S0.v = __builtin_amdgcn_wmma_f32_16x16x32_bf16(false, aq[f].v, false, bk.v,
                                                     (short)0, S0.v, false, false);
      int s2 = kt + 16 + r16; if (s2 > TCTX - 1) s2 = TCTX - 1;  // clamp (masked anyway)
      const bfu* krow2 = kb + ((size_t)(b * TCTX + s2)) * E_SZ + h * HD_SZ;
      bk.q[0] = *(const uint4*)(krow2 + f * 32 + ka);
      bk.q[1] = *(const uint4*)(krow2 + f * 32 + ka + 16);
      S1.v = __builtin_amdgcn_wmma_f32_16x16x32_bf16(false, aq[f].v, false, bk.v,
                                                     (short)0, S1.v, false, false);
    }
    // online softmax per row (j + 8*hf); cols across the 16-lane half
    for (int j = 0; j < 8; j++) {
      int qa = q0 + j + 8 * hf;
      float s0 = S0.f[j] * scale; if (kt + r16 > qa)      s0 = -1e30f;
      float s1 = S1.f[j] * scale; if (kt + 16 + r16 > qa) s1 = -1e30f;
      float mt = fmaxf(s0, s1);
      for (int o = 1; o < 16; o <<= 1) mt = fmaxf(mt, __shfl_xor(mt, o, 16));
      float mnew = fmaxf(mrow[j], mt);
      float corr = __expf(mrow[j] - mnew);
      float p0 = __expf(s0 - mnew), p1 = __expf(s1 - mnew);
      float ps = p0 + p1;
      for (int o = 1; o < 16; o <<= 1) ps += __shfl_xor(ps, o, 16);
      lrow[j] = lrow[j] * corr + ps;
      for (int n = 0; n < 4; n++) O[n].f[j] *= corr;
      mrow[j] = mnew;
      int prow = j + 8 * hf;
      Pbe[prow * 32 + r16]      = f2bfu(p0);
      Pbe[prow * 32 + 16 + r16] = f2bfu(p1);
    }
    asm volatile("s_wait_dscnt 0" ::: "memory");  // LDS transpose fence (wave-private)
    __builtin_amdgcn_wave_barrier();
    FragA pF;
    pF.q[0] = Pb4[wave][r16 * 4 + (ka >> 3)];
    pF.q[1] = Pb4[wave][r16 * 4 + (ka >> 3) + 2];
    __builtin_amdgcn_wave_barrier();
    for (int n = 0; n < 4; n++) {
      FragA vF;
      int d = n * 16 + r16;
      for (int i = 0; i < 8; i++) {
        int s  = kt + ka + i;       if (s  > TCTX - 1) s  = TCTX - 1;
        int s2 = kt + 16 + ka + i;  if (s2 > TCTX - 1) s2 = TCTX - 1;
        vF.u[i]     = vb[((size_t)(b * TCTX + s))  * E_SZ + h * HD_SZ + d];
        vF.u[8 + i] = vb[((size_t)(b * TCTX + s2)) * E_SZ + h * HD_SZ + d];
      }
      O[n].v = __builtin_amdgcn_wmma_f32_16x16x32_bf16(false, pF.v, false, vF.v,
                                                       (short)0, O[n].v, false, false);
    }
  }
  for (int j = 0; j < 8; j++) {
    float inv = 1.0f / lrow[j];
    int t = q0 + j + 8 * hf;
    for (int n = 0; n < 4; n++)
      ab[((size_t)(b * TCTX + t)) * E_SZ + h * HD_SZ + n * 16 + r16] = f2bfu(O[n].f[j] * inv);
  }
}

// ---------------------------------------------------------------- loss
__global__ void zero_loss_kernel(float* loss) { if (threadIdx.x == 0) *loss = 0.0f; }

__global__ void loss_kernel(const float* __restrict__ logits, const int* __restrict__ targets,
                            float* __restrict__ loss) {
  __shared__ float red[256];
  int r = blockIdx.x, tid = threadIdx.x;
  const float* row = logits + (size_t)r * V_SZ;
  float mx = -1e30f;
  for (int i = tid; i < V_SZ; i += 256) mx = fmaxf(mx, row[i]);
  red[tid] = mx; __syncthreads();
  for (int o = 128; o > 0; o >>= 1) { if (tid < o) red[tid] = fmaxf(red[tid], red[tid + o]); __syncthreads(); }
  mx = red[0]; __syncthreads();
  float s = 0.f;
  for (int i = tid; i < V_SZ; i += 256) s += __expf(row[i] - mx);
  red[tid] = s; __syncthreads();
  for (int o = 128; o > 0; o >>= 1) { if (tid < o) red[tid] += red[tid + o]; __syncthreads(); }
  if (tid == 0) {
    float lse = mx + __logf(red[0]);
    atomicAdd(loss, (lse - row[targets[r]]) * (1.0f / (float)BT));
  }
}

// ---------------------------------------------------------------- driver
extern "C" void kernel_launch(void* const* d_in, const int* in_sizes, int n_in,
                              void* d_out, int out_size, void* d_ws, size_t ws_size,
                              hipStream_t stream) {
  (void)in_sizes; (void)n_in; (void)out_size; (void)ws_size;
  const int*   idx     = (const int*)  d_in[0];
  const int*   targets = (const int*)  d_in[1];
  const float* tok     = (const float*)d_in[2];
  const float* pos     = (const float*)d_in[3];
  const float* Wq      = (const float*)d_in[4];
  const float* Wk      = (const float*)d_in[5];
  const float* Wv      = (const float*)d_in[6];
  const float* Wproj   = (const float*)d_in[7];
  const float* bproj   = (const float*)d_in[8];
  const float* ln1_g   = (const float*)d_in[9];
  const float* ln1_b   = (const float*)d_in[10];
  const float* ln2_g   = (const float*)d_in[11];
  const float* ln2_b   = (const float*)d_in[12];
  const float* W1      = (const float*)d_in[13];
  const float* b1      = (const float*)d_in[14];
  const float* W2      = (const float*)d_in[15];
  const float* b2      = (const float*)d_in[16];
  const float* lnf_g   = (const float*)d_in[17];
  const float* lnf_b   = (const float*)d_in[18];
  const float* Whead   = (const float*)d_in[19];
  const float* bhead   = (const float*)d_in[20];

  char* ws = (char*)d_ws;
  const size_t MB = 1024 * 1024;
  float* x  = (float*)(ws + 0);        // 16 MB  [BT,E] f32 residual stream
  bfu*   h  = (bfu*)(ws + 16 * MB);    //  8 MB  LN output bf16
  bfu*   qb = (bfu*)(ws + 24 * MB);    //  8 MB
  bfu*   kb = (bfu*)(ws + 32 * MB);    //  8 MB
  bfu*   vb = (bfu*)(ws + 40 * MB);    //  8 MB
  bfu*   ab = (bfu*)(ws + 48 * MB);    //  8 MB attention out (head-concat)
  bfu*   fb = (bfu*)(ws + 56 * MB);    // 32 MB MLP hidden bf16
  bfu*   wb = (bfu*)(ws + 88 * MB);    // 65.5 MB packed transposed-weight scratch (reused)

  float* logits = (float*)d_out;               // [BT, V]
  float* loss   = logits + (size_t)BT * V_SZ;  // scalar after logits

  dim3 blk(256);
  dim3 gE(E_SZ / 128, BT / 128);     // (8,32)
  dim3 gF(4 * E_SZ / 128, BT / 128); // (32,32)
  dim3 gH(V_SZ / 128, BT / 128);     // (250,32)

  embed_kernel<<<BT, blk, 0, stream>>>(idx, tok, pos, x);

  for (int l = 0; l < L_N; l++) {
    const size_t wEE = (size_t)E_SZ * E_SZ;
    ln_kernel<<<BT, blk, 0, stream>>>(x, ln1_g + l * E_SZ, ln1_b + l * E_SZ, h);

    pack_qkvT_kernel<<<E_SZ * E_SZ / 256, blk, 0, stream>>>(Wq + (size_t)l * wEE, wb);
    gemm_bf16_kernel<false, false, false, true><<<gE, blk, 0, stream>>>(
        h, wb, nullptr, nullptr, qb, BT, E_SZ, E_SZ);
    pack_qkvT_kernel<<<E_SZ * E_SZ / 256, blk, 0, stream>>>(Wk + (size_t)l * wEE, wb);
    gemm_bf16_kernel<false, false, false, true><<<gE, blk, 0, stream>>>(
        h, wb, nullptr, nullptr, kb, BT, E_SZ, E_SZ);
    pack_qkvT_kernel<<<E_SZ * E_SZ / 256, blk, 0, stream>>>(Wv + (size_t)l * wEE, wb);
    gemm_bf16_kernel<false, false, false, true><<<gE, blk, 0, stream>>>(
        h, wb, nullptr, nullptr, vb, BT, E_SZ, E_SZ);

    attention_kernel<<<BATCH * H_N * (TCTX / 128), blk, 0, stream>>>(qb, kb, vb, ab);

    pack_wT_kernel<<<4096, blk, 0, stream>>>(Wproj + (size_t)l * wEE, wb, E_SZ, E_SZ);
    gemm_bf16_kernel<true, false, true, false><<<gE, blk, 0, stream>>>(
        ab, wb, bproj + (size_t)l * E_SZ, x, x, BT, E_SZ, E_SZ);

    ln_kernel<<<BT, blk, 0, stream>>>(x, ln2_g + l * E_SZ, ln2_b + l * E_SZ, h);

    pack_wT_kernel<<<4096, blk, 0, stream>>>(W1 + (size_t)l * wEE * 4, wb, E_SZ, 4 * E_SZ);
    gemm_bf16_kernel<true, true, false, true><<<gF, blk, 0, stream>>>(
        h, wb, b1 + (size_t)l * 4 * E_SZ, nullptr, fb, BT, 4 * E_SZ, E_SZ);

    pack_wT_kernel<<<4096, blk, 0, stream>>>(W2 + (size_t)l * wEE * 4, wb, 4 * E_SZ, E_SZ);
    gemm_bf16_kernel<true, false, true, false><<<gE, blk, 0, stream>>>(
        fb, wb, b2 + (size_t)l * E_SZ, x, x, BT, E_SZ, 4 * E_SZ);
  }

  ln_kernel<<<BT, blk, 0, stream>>>(x, lnf_g, lnf_b, h);
  pack_wT_kernel<<<4096, blk, 0, stream>>>(Whead, wb, E_SZ, V_SZ);
  gemm_bf16_kernel<true, false, false, false><<<gH, blk, 0, stream>>>(
      h, wb, bhead, nullptr, logits, BT, V_SZ, E_SZ);

  zero_loss_kernel<<<1, 32, 0, stream>>>(loss);
  loss_kernel<<<BT, blk, 0, stream>>>(logits, targets, loss);
}
